// HBitLinear_87162066305077
// MI455X (gfx1250) — compile-verified, matrix operand order checked by reading, not run on hardware
//
#include <hip/hip_runtime.h>

// ---------------------------------------------------------------------------
// HBitLinear forward for MI455X (gfx1250):
//   out = FWHT( quant4( FWHT(x) ) ) @ ternary(W)^T + bias
// FWHT done in LDS (O(N log N)); GEMM via v_wmma_f32_16x16x32_bf16 with
// GLOBAL_LOAD_ASYNC_TO_LDS double-buffered staging (ASYNCcnt path).
// ---------------------------------------------------------------------------

#define DD 4096          // feature dim
#define MMROWS 8192      // B*S = 4*2048
#define LSTR 40          // LDS row stride in halfs (32 data + 8 pad)

typedef __attribute__((ext_vector_type(16))) __bf16 v16bf;
typedef __attribute__((ext_vector_type(8)))  float  v8f;
typedef int v4i_g __attribute__((vector_size(16)));   // GCC-style int4 (builtin param type)

#if __has_builtin(__builtin_amdgcn_global_load_async_to_lds_b128)
#define USE_ASYNC_LDS 1
#else
#define USE_ASYNC_LDS 0
#endif

static __device__ __forceinline__ void async_ld_b128(const void* g, void* l) {
#if USE_ASYNC_LDS
  // global as(1) pointer: flat == global numerically.
  // lds as(3) pointer: low 32 bits of the flat shared address ARE the LDS
  // offset on CDNA5 (aperture in addr[63:32]).
  __builtin_amdgcn_global_load_async_to_lds_b128(
      (__attribute__((address_space(1))) v4i_g*)(unsigned long long)g,
      (__attribute__((address_space(3))) v4i_g*)(unsigned int)(unsigned long long)l,
      0, 0);
#else
  *(uint4*)l = *(const uint4*)g;   // sync fallback: load + ds_store
#endif
}

static __device__ __forceinline__ void wait_async0() {
#if USE_ASYNC_LDS
#if __has_builtin(__builtin_amdgcn_s_wait_asynccnt)
  __builtin_amdgcn_s_wait_asynccnt(0);
#else
  asm volatile("s_wait_asynccnt 0" ::: "memory");
#endif
#endif
}

static __device__ __forceinline__ unsigned short f2bf(float f) {
  unsigned int u = __builtin_bit_cast(unsigned int, f);
  unsigned int lsb = (u >> 16) & 1u;
  u += 0x7fffu + lsb;                 // round-to-nearest-even
  return (unsigned short)(u >> 16);
}

// In-LDS fast Walsh-Hadamard transform of 4096 floats, 256 threads.
static __device__ __forceinline__ void fwht4096(float* u, int tid) {
  for (int s = 0; s < 12; ++s) {
    const int h = 1 << s;
    __syncthreads();
    for (int idx = tid; idx < 2048; idx += 256) {
      const int i = ((idx >> s) << (s + 1)) + (idx & (h - 1));
      const int j = i + h;
      const float a = u[i], b = u[j];
      u[i] = a + b;
      u[j] = a - b;
    }
  }
  __syncthreads();
}

// ------------------------- weight quantization -----------------------------

__global__ __launch_bounds__(256) void k_wabs(const float* __restrict__ w,
                                              float* __restrict__ pw) {
  __shared__ float r[256];
  const int tid = threadIdx.x;
  const float4* w4 = (const float4*)w;
  float m = 0.0f;
  for (size_t i = (size_t)blockIdx.x * 256 + tid; i < (size_t)4194304;
       i += (size_t)gridDim.x * 256) {
    const float4 v = w4[i];
    m = fmaxf(m, fmaxf(fmaxf(fabsf(v.x), fabsf(v.y)),
                       fmaxf(fabsf(v.z), fabsf(v.w))));
  }
  r[tid] = m;
  __syncthreads();
  for (int s = 128; s > 0; s >>= 1) {
    if (tid < s) r[tid] = fmaxf(r[tid], r[tid + s]);
    __syncthreads();
  }
  if (tid == 0) pw[blockIdx.x] = r[0];
}

__global__ __launch_bounds__(256) void k_wscale(const float* __restrict__ pw,
                                                float* __restrict__ scal) {
  __shared__ float r[256];
  const int tid = threadIdx.x;
  r[tid] = fmaxf(pw[tid], pw[tid + 256]);
  __syncthreads();
  for (int s = 128; s > 0; s >>= 1) {
    if (tid < s) r[tid] = fmaxf(r[tid], r[tid + s]);
    __syncthreads();
  }
  if (tid == 0) scal[0] = r[0];
}

static __device__ __forceinline__ float tern1(float wv, float s) {
  // q = sign(w/s)*s, zeroed where |w/s| < 0.5 ; s==0 => all weights zero
  return (s > 0.0f && (2.0f * fabsf(wv) >= s)) ? copysignf(s, wv) : 0.0f;
}

__global__ __launch_bounds__(256) void k_wquant(const float* __restrict__ w,
                                                const float* __restrict__ scal,
                                                unsigned short* __restrict__ wq) {
  const float s = scal[0];
  const float4* w4 = (const float4*)w;
  uint2* o2 = (uint2*)wq;
  for (size_t i = (size_t)blockIdx.x * 256 + threadIdx.x; i < (size_t)4194304;
       i += (size_t)gridDim.x * 256) {
    const float4 v = w4[i];
    const unsigned short b0 = f2bf(tern1(v.x, s));
    const unsigned short b1 = f2bf(tern1(v.y, s));
    const unsigned short b2 = f2bf(tern1(v.z, s));
    const unsigned short b3 = f2bf(tern1(v.w, s));
    uint2 o;
    o.x = (unsigned int)b0 | ((unsigned int)b1 << 16);
    o.y = (unsigned int)b2 | ((unsigned int)b3 << 16);
    o2[i] = o;
  }
}

// --------------------- activation: FWHT + 4-bit quant ----------------------

__global__ __launch_bounds__(256) void k_fwht1_minmax(const float* __restrict__ x,
                                                      float* __restrict__ t,
                                                      float* __restrict__ pmin,
                                                      float* __restrict__ pmax) {
  __shared__ float u[4096];
  __shared__ float rmn[256], rmx[256];
  const int tid = threadIdx.x;
  const size_t row = blockIdx.x;
  const float4* xr = (const float4*)(x + row * (size_t)DD);
  for (int c = tid; c < 1024; c += 256) {
    const float4 v = xr[c];
    u[c * 4 + 0] = v.x; u[c * 4 + 1] = v.y;
    u[c * 4 + 2] = v.z; u[c * 4 + 3] = v.w;
  }
  fwht4096(u, tid);                      // leading syncthreads covers the load
  const float nrm = 0.015625f;           // (1/sqrt2)^12 = 2^-6
  float4* tr = (float4*)(t + row * (size_t)DD);
  float lmn = 3.4e38f, lmx = -3.4e38f;
  for (int c = tid; c < 1024; c += 256) {
    float4 v;
    v.x = u[c * 4 + 0] * nrm; v.y = u[c * 4 + 1] * nrm;
    v.z = u[c * 4 + 2] * nrm; v.w = u[c * 4 + 3] * nrm;
    lmn = fminf(lmn, fminf(fminf(v.x, v.y), fminf(v.z, v.w)));
    lmx = fmaxf(lmx, fmaxf(fmaxf(v.x, v.y), fmaxf(v.z, v.w)));
    tr[c] = v;
  }
  rmn[tid] = lmn; rmx[tid] = lmx;
  __syncthreads();
  for (int s = 128; s > 0; s >>= 1) {
    if (tid < s) {
      rmn[tid] = fminf(rmn[tid], rmn[tid + s]);
      rmx[tid] = fmaxf(rmx[tid], rmx[tid + s]);
    }
    __syncthreads();
  }
  if (tid == 0) { pmin[blockIdx.x] = rmn[0]; pmax[blockIdx.x] = rmx[0]; }
}

__global__ __launch_bounds__(256) void k_qparams(const float* __restrict__ pmin,
                                                 const float* __restrict__ pmax,
                                                 float* __restrict__ scal) {
  __shared__ float rmn[256], rmx[256];
  const int tid = threadIdx.x;
  float lmn = 3.4e38f, lmx = -3.4e38f;
  for (int i = tid; i < MMROWS; i += 256) {
    lmn = fminf(lmn, pmin[i]);
    lmx = fmaxf(lmx, pmax[i]);
  }
  rmn[tid] = lmn; rmx[tid] = lmx;
  __syncthreads();
  for (int s = 128; s > 0; s >>= 1) {
    if (tid < s) {
      rmn[tid] = fminf(rmn[tid], rmn[tid + s]);
      rmx[tid] = fmaxf(rmx[tid], rmx[tid + s]);
    }
    __syncthreads();
  }
  if (tid == 0) {
    float sc = (rmx[0] - rmn[0]) / 15.0f;   // (qmax-qmin) = 15 for 4-bit
    if (!(sc > 0.0f)) sc = 1.0f;            // degenerate guard
    const float zp = -8.0f - rintf(rmn[0] / sc);
    scal[1] = sc;
    scal[2] = zp;
  }
}

__global__ __launch_bounds__(256) void k_fwht2_q(const float* __restrict__ t,
                                                 const float* __restrict__ scal,
                                                 unsigned short* __restrict__ xq) {
  __shared__ float u[4096];
  const int tid = threadIdx.x;
  const size_t row = blockIdx.x;
  const float sc = scal[1], zp = scal[2];
  const float4* tr = (const float4*)(t + row * (size_t)DD);
  for (int c = tid; c < 1024; c += 256) {
    const float4 v = tr[c];
    u[c * 4 + 0] = sc * (fminf(fmaxf(rintf(v.x / sc + zp), -8.0f), 7.0f) - zp);
    u[c * 4 + 1] = sc * (fminf(fmaxf(rintf(v.y / sc + zp), -8.0f), 7.0f) - zp);
    u[c * 4 + 2] = sc * (fminf(fmaxf(rintf(v.z / sc + zp), -8.0f), 7.0f) - zp);
    u[c * 4 + 3] = sc * (fminf(fmaxf(rintf(v.w / sc + zp), -8.0f), 7.0f) - zp);
  }
  fwht4096(u, tid);
  const float nrm = 0.015625f;
  unsigned int* xo = (unsigned int*)(xq + row * (size_t)DD);
  for (int c = tid; c < 2048; c += 256) {
    const unsigned short lo = f2bf(u[c * 2 + 0] * nrm);
    const unsigned short hi = f2bf(u[c * 2 + 1] * nrm);
    xo[c] = (unsigned int)lo | ((unsigned int)hi << 16);
  }
}

// ------------------------------- WMMA GEMM ---------------------------------
// out[M=8192, N=4096] = A[M,K] * Bw[N,K]^T + bias  (A,Bw bf16, out f32)
// Block tile 128(M) x 256(N), 8 waves as 2x4, 64x64 per wave (16 wmma/K-step).

static __device__ __forceinline__ v16bf load_frag(const unsigned short* tile,
                                                  int rowBase, int lane) {
  // 16-bit operand lane layout (ISA 7.12.2): lanes 0-15 hold rows 0..15,
  // VGPR j holds K pair p = (j<4) ? g*4+j : 8+g*4+(j-4), where g = lane/16.
  const int l16 = lane & 15;
  const int g = (lane >> 4) & 1;
  const unsigned int* rp = (const unsigned int*)(tile + (rowBase + l16) * LSTR);
  union { v16bf v; unsigned int u[8]; } f;
#pragma unroll
  for (int j = 0; j < 8; ++j) {
    const int p = (j < 4) ? (g * 4 + j) : (8 + g * 4 + (j - 4));
    f.u[j] = rp[p];
  }
  return f.v;
}

__global__ __launch_bounds__(256) void k_gemm(const unsigned short* __restrict__ A,
                                              const unsigned short* __restrict__ Bw,
                                              const float* __restrict__ bias,
                                              float* __restrict__ out) {
  __shared__ unsigned short As[2][128 * LSTR];
  __shared__ unsigned short Bs[2][256 * LSTR];

  const int tid  = threadIdx.x;
  const int lane = tid & 31;
  const int wave = tid >> 5;      // 8 waves
  const int wm   = wave >> 2;     // 0..1  (64-row slabs)
  const int wn   = wave & 3;      // 0..3  (64-col slabs)
  const int m0   = blockIdx.y * 128;
  const int n0   = blockIdx.x * 256;

  // Async staging: A tile 128x32 halfs = 512 b128 chunks (2/thread),
  //                B tile 256x32 halfs = 1024 b128 chunks (4/thread).
  const int rA0 = tid >> 2,          qA0 = (tid & 3) * 8;
  const int rA1 = (tid + 256) >> 2,  qA1 = qA0;          // (tid+256)&3 == tid&3

  v8f acc[4][4];
#pragma unroll
  for (int i = 0; i < 4; ++i)
#pragma unroll
    for (int j = 0; j < 4; ++j) {
      v8f z = {0.f, 0.f, 0.f, 0.f, 0.f, 0.f, 0.f, 0.f};
      acc[i][j] = z;
    }

  const unsigned short* Abase = A  + (size_t)m0 * DD;
  const unsigned short* Bbase = Bw + (size_t)n0 * DD;

  // stage K-tile (k0) into buffer `buf`
  auto stage = [&](int k0, int buf) {
    async_ld_b128(Abase + (size_t)rA0 * DD + k0 + qA0, &As[buf][rA0 * LSTR + qA0]);
    async_ld_b128(Abase + (size_t)rA1 * DD + k0 + qA1, &As[buf][rA1 * LSTR + qA1]);
#pragma unroll
    for (int tchunk = 0; tchunk < 4; ++tchunk) {
      const int c  = tid + 256 * tchunk;
      const int rB = c >> 2, qB = (c & 3) * 8;
      async_ld_b128(Bbase + (size_t)rB * DD + k0 + qB, &Bs[buf][rB * LSTR + qB]);
    }
  };

  stage(0, 0);
  wait_async0();
  __syncthreads();

  for (int kk = 0; kk < 128; ++kk) {
    const int cur = kk & 1;
    if (kk < 127) {
      stage((kk + 1) * 32, cur ^ 1);
      if (kk < 126) {  // pull the K-tile after next into L2 (global_prefetch_b8)
        __builtin_prefetch(Abase + (size_t)rA0 * DD + (kk + 2) * 32, 0, 1);
        __builtin_prefetch(Bbase + (size_t)rA0 * DD + (kk + 2) * 32, 0, 1);
      }
    }

    v16bf af[4], bf[4];
#pragma unroll
    for (int i = 0; i < 4; ++i) af[i] = load_frag(As[cur], wm * 64 + i * 16, lane);
#pragma unroll
    for (int j = 0; j < 4; ++j) bf[j] = load_frag(Bs[cur], wn * 64 + j * 16, lane);

#pragma unroll
    for (int i = 0; i < 4; ++i)
#pragma unroll
      for (int j = 0; j < 4; ++j)
        acc[i][j] = __builtin_amdgcn_wmma_f32_16x16x32_bf16(
            false, af[i], false, bf[j], (short)0, acc[i][j], false, false);

    wait_async0();       // next buffer's async writes landed in LDS
    __syncthreads();     // ... and every wave is done reading `cur`
  }

  // Epilogue: C/D layout (ISA 7.12.2): VGPR r, lanes 0-15 -> M=base+r,
  // lanes 16-31 -> M=base+8+r; N = lane%16.
  const int l16 = lane & 15;
  const int hi  = (lane >> 4) * 8;
#pragma unroll
  for (int i = 0; i < 4; ++i) {
#pragma unroll
    for (int j = 0; j < 4; ++j) {
      const int mbase = m0 + wm * 64 + i * 16 + hi;
      const int nn    = n0 + wn * 64 + j * 16 + l16;
      const float bv  = bias[nn];
#pragma unroll
      for (int r = 0; r < 8; ++r)
        out[(size_t)(mbase + r) * DD + nn] = acc[i][j][r] + bv;
    }
  }
}

// ------------------------------- launcher ----------------------------------

extern "C" void kernel_launch(void* const* d_in, const int* in_sizes, int n_in,
                              void* d_out, int out_size, void* d_ws, size_t ws_size,
                              hipStream_t stream) {
  const float* x    = (const float*)d_in[0];   // (4,2048,4096) f32
  const float* w    = (const float*)d_in[1];   // (4096,4096)   f32
  const float* bias = (const float*)d_in[2];   // (4096,)       f32
  float* out = (float*)d_out;                  // (4,2048,4096) f32

  char* ws = (char*)d_ws;
  unsigned short* xq = (unsigned short*)(ws + 0);           // 64 MiB bf16 activations
  unsigned short* wq = (unsigned short*)(ws + 67108864);    // 32 MiB bf16 ternary W
  float* pw   = (float*)(ws + 100663296);                   // 512 partial |W| maxima
  float* pmin = (float*)(ws + 100663296 + 4096);            // 8192 partial mins
  float* pmax = (float*)(ws + 100663296 + 4096 + 32768);    // 8192 partial maxs
  float* scal = (float*)(ws + 100663296 + 4096 + 65536);    // [wscale, qscale, qzp]

  float* t = out;  // stage FWHT(x) in d_out (exact size), overwritten by GEMM

  k_wabs        <<<512,           256, 0, stream>>>(w, pw);
  k_wscale      <<<1,             256, 0, stream>>>(pw, scal);
  k_wquant      <<<4096,          256, 0, stream>>>(w, scal, wq);
  k_fwht1_minmax<<<MMROWS,        256, 0, stream>>>(x, t, pmin, pmax);
  k_qparams     <<<1,             256, 0, stream>>>(pmin, pmax, scal);
  k_fwht2_q     <<<MMROWS,        256, 0, stream>>>(t, scal, xq);
  k_gemm        <<<dim3(16, 64),  256, 0, stream>>>(xq, wq, bias, out);
}
